// InvariantPointAttention_90460601188764
// MI455X (gfx1250) — matrix operand order repaired and checked by reading, not compiled, not run on hardware
//
#include <hip/hip_runtime.h>
#include <hip/hip_bf16.h>
#include <math.h>

typedef __bf16 bf16;
typedef __attribute__((ext_vector_type(16))) __bf16 bf16x16;
typedef __attribute__((ext_vector_type(8)))  float  f32x8;
typedef __attribute__((ext_vector_type(4)))  int    i32x4;

union FragCast { bf16x16 v; i32x4 q[2]; };

#define NSEQ   2048
#define BATCH  2
#define NHEADS 8
#define ROWS   (BATCH*NSEQ)          /* 4096 */
#define NCOLS_PROJ 1056              /* 3*256 + 3*96 */
#define QK_LD  64                    /* 44 used, padded to 64 (K dim of S)   */
#define V_CD   48                    /* 44 used, padded to 48 (N dim of P*V) */
#define CAT_LD 352                   /* 256 + 96 concat width */

#define WMMA_BF16(a,b,c) __builtin_amdgcn_wmma_f32_16x16x32_bf16(false,(a),false,(b),(short)0,(c),false,false)

__device__ __forceinline__ float softplus_f(float x) { return logf(1.0f + __expf(x)); }

// A fragment (16x32 bf16, MxK) from row-major [row][k], ld elements per row.
// Lane layout: lanes 0-15 row m=lane, k = {0..7, 16..23}; lanes 16-31 k = {8..15, 24..31}.
__device__ __forceinline__ bf16x16 load_frag_a(const bf16* base, int ld, int row0, int k0) {
  const int lane = threadIdx.x & 31;
  const int hh = lane >> 4, m = lane & 15;
  const bf16* p = base + (size_t)(row0 + m) * ld + k0 + 8 * hh;
  FragCast u;
  u.q[0] = *(const i32x4*)(p);
  u.q[1] = *(const i32x4*)(p + 16);
  return u.v;
}

// B fragment (32x16 bf16, KxN) from a [col][k] matrix (i.e. B stored transposed).
// Lane layout: lane -> column n=lane&15; lanes 0-15 k = {0..7, 16..23}; lanes 16-31 k = {8..15, 24..31}.
__device__ __forceinline__ bf16x16 load_frag_b(const bf16* base, int ld, int col0, int k0) {
  const int lane = threadIdx.x & 31;
  const int hh = lane >> 4, n = lane & 15;
  const bf16* p = base + (size_t)(col0 + n) * ld + k0 + 8 * hh;
  FragCast u;
  u.q[0] = *(const i32x4*)(p);
  u.q[1] = *(const i32x4*)(p + 16);
  return u.v;
}

/* ---------------- prep kernels ---------------- */

__global__ void k_prep_feat(const float* f, bf16* out, int nelem) {
  int i = blockIdx.x * blockDim.x + threadIdx.x;
  if (i < nelem) out[i] = (bf16)f[i];
}

__global__ void k_prep_wcat(const float* Wq, const float* Wk, const float* Wv,
                            const float* Wqp, const float* Wkp, const float* Wvp,
                            bf16* WT) {
  int idx = blockIdx.x * blockDim.x + threadIdx.x;
  if (idx >= NCOLS_PROJ * 256) return;
  int o = idx >> 8;       // output column 0..1055
  int k = idx & 255;      // input dim
  float v;
  if (o < 256)       v = Wq [(size_t)k * 256 + o];
  else if (o < 512)  v = Wk [(size_t)k * 256 + (o - 256)];
  else if (o < 768)  v = Wv [(size_t)k * 256 + (o - 512)];
  else if (o < 864)  v = Wqp[(size_t)k * 96  + (o - 768)];
  else if (o < 960)  v = Wkp[(size_t)k * 96  + (o - 864)];
  else               v = Wvp[(size_t)k * 96  + (o - 960)];
  WT[(size_t)o * 256 + k] = (bf16)v;
}

__global__ void k_prep_woT(const float* Wo, bf16* WoT) {
  int idx = blockIdx.x * blockDim.x + threadIdx.x;
  if (idx >= 256 * CAT_LD) return;
  int n = idx / CAT_LD, k = idx % CAT_LD;
  WoT[(size_t)n * CAT_LD + k] = (bf16)Wo[(size_t)k * 256 + n];
}

/* ---------------- fused projection GEMM: X(4096x256) @ Wcat(256x1056) ---------------- */

__global__ __launch_bounds__(32) void k_gemm_proj(
    const bf16* __restrict__ Fbf, const bf16* __restrict__ WT,
    const float* bq, const float* bk, const float* bv,
    const float* bqp, const float* bkp, const float* bvp,
    const float* coords, const float* w_l, const float* w_c,
    bf16* Qhat, bf16* Khat, bf16* VhatT) {
  const int row0 = blockIdx.x * 16;
  const int col0 = blockIdx.y * 32;
  const int lane = threadIdx.x & 31;
  const int hh = lane >> 4, n16 = lane & 15;

  f32x8 c0 = {0,0,0,0,0,0,0,0}, c1 = {0,0,0,0,0,0,0,0};
  for (int k0 = 0; k0 < 256; k0 += 32) {
    bf16x16 a  = load_frag_a(Fbf, 256, row0, k0);
    bf16x16 b0 = load_frag_b(WT, 256, col0, k0);
    bf16x16 b1 = load_frag_b(WT, 256, col0 + 16, k0);
    c0 = WMMA_BF16(a, b0, c0);
    c1 = WMMA_BF16(a, b1, c1);
  }
  #pragma unroll
  for (int t = 0; t < 2; ++t) {
    f32x8 c = t ? c1 : c0;
    const int o = col0 + 16 * t + n16;       // global output column, fixed per lane
    #pragma unroll
    for (int r = 0; r < 8; ++r) {
      int g = row0 + r + 8 * hh;             // global row in [0,4096)
      int bidx = g >> 11, i = g & 2047;
      float val = c[r];
      if (o < 768) {                         // q / k / v blocks
        int which = o >> 8; int d = o & 255; int hd = d >> 5; int dd = d & 31;
        size_t rowix = (size_t)(bidx * NHEADS + hd) * NSEQ + i;
        if (which == 0) {
          float wl = softplus_f(w_l[hd]) * 0.17677669529663689f;  // softplus(wl)/sqrt(32)
          Qhat[rowix * QK_LD + dd] = (bf16)((val + bq[d]) * wl);
        } else if (which == 1) {
          Khat[rowix * QK_LD + dd] = (bf16)(val + bk[d]);
        } else {
          VhatT[((size_t)(bidx * NHEADS + hd) * V_CD + dd) * NSEQ + i] = (bf16)(val + bv[d]);
        }
      } else {                               // qp / kp / vp blocks (+ coords)
        int o2 = o - 768; int which = o2 / 96; int d = o2 % 96;
        int hd = d / 12; int rem = d % 12; int axis = rem % 3;
        float cv = coords[((size_t)bidx * NSEQ + i) * 3 + axis];
        size_t rowix = (size_t)(bidx * NHEADS + hd) * NSEQ + i;
        if (which == 0) {
          float wc = softplus_f(w_c[hd]);
          Qhat[rowix * QK_LD + 32 + rem] = (bf16)((val + bqp[d] + cv) * wc);
        } else if (which == 1) {
          Khat[rowix * QK_LD + 32 + rem] = (bf16)(val + bkp[d] + cv);
        } else {
          VhatT[((size_t)(bidx * NHEADS + hd) * V_CD + 32 + rem) * NSEQ + i] = (bf16)(val + bvp[d] + cv);
        }
      }
    }
  }
}

/* ---------------- per-key bias: -0.5*softplus(w_c[h])*||kp_j||^2 ---------------- */

__global__ void k_kk_bias(const bf16* Khat, const float* w_c, float* biasK) {
  int idx = blockIdx.x * blockDim.x + threadIdx.x;
  if (idx >= BATCH * NHEADS * NSEQ) return;
  int bh = idx / NSEQ; int hd = bh & 7;
  const bf16* kp = Khat + (size_t)idx * QK_LD + 32;
  float s = 0.f;
  #pragma unroll
  for (int d = 0; d < 12; ++d) { float x = (float)kp[d]; s += x * x; }
  biasK[idx] = -0.5f * softplus_f(w_c[hd]) * s;
}

/* ---------------- flash attention per (b,h): 16 query rows / wave ---------------- */

__global__ __launch_bounds__(32) void k_attn(
    const bf16* __restrict__ Qhat, const bf16* __restrict__ Khat,
    const bf16* __restrict__ VhatT, const float* __restrict__ biasK,
    const unsigned char* __restrict__ mask, bf16* attcat) {
  const int bh = blockIdx.x;            // b*8 + h
  const int bidx = bh >> 3, hd = bh & 7;
  const int i0 = blockIdx.y * 16;
  const bf16* Q = Qhat + (size_t)bh * NSEQ * QK_LD;
  const bf16* K = Khat + (size_t)bh * NSEQ * QK_LD;
  const bf16* V = VhatT + (size_t)bh * V_CD * NSEQ;
  const float* bias = biasK + (size_t)bh * NSEQ;
  const unsigned char* mrow = mask + ((size_t)bidx * NSEQ + i0) * NSEQ;

  const int lane = threadIdx.x & 31;
  const int hh = lane >> 4, n16 = lane & 15;

  const bf16x16 aq0 = load_frag_a(Q, QK_LD, i0, 0);
  const bf16x16 aq1 = load_frag_a(Q, QK_LD, i0, 32);

  f32x8 o0 = {0,0,0,0,0,0,0,0}, o1 = {0,0,0,0,0,0,0,0}, o2 = {0,0,0,0,0,0,0,0};
  float mst[8], lst[8];
  #pragma unroll
  for (int r = 0; r < 8; ++r) { mst[r] = -1e30f; lst[r] = 0.f; }

  __shared__ __align__(16) bf16 pls[16 * 32];

  for (int j0 = 0; j0 < NSEQ; j0 += 32) {
    // S tiles: S = Qhat @ Khat^T  (K=64 padded, two j sub-tiles of 16)
    f32x8 s0 = {0,0,0,0,0,0,0,0}, s1 = {0,0,0,0,0,0,0,0};
    {
      bf16x16 b0 = load_frag_b(K, QK_LD, j0, 0);
      bf16x16 b1 = load_frag_b(K, QK_LD, j0, 32);
      s0 = WMMA_BF16(aq0, b0, s0);
      s0 = WMMA_BF16(aq1, b1, s0);
      bf16x16 b2 = load_frag_b(K, QK_LD, j0 + 16, 0);
      bf16x16 b3 = load_frag_b(K, QK_LD, j0 + 16, 32);
      s1 = WMMA_BF16(aq0, b2, s1);
      s1 = WMMA_BF16(aq1, b3, s1);
    }
    const float bj0 = bias[j0 + n16];
    const float bj1 = bias[j0 + 16 + n16];
    #pragma unroll
    for (int r = 0; r < 8; ++r) {
      float v0 = s0[r] + bj0;
      float v1 = s1[r] + bj1;
      const unsigned char* mp = mrow + (size_t)(r + 8 * hh) * NSEQ + j0;
      if (mp[n16])      v0 = -1e30f;
      if (mp[16 + n16]) v1 = -1e30f;
      s0[r] = v0; s1[r] = v1;
    }
    // row max across the 16 lanes of this half-wave (one row's 16 columns live there)
    float tmax[8];
    #pragma unroll
    for (int r = 0; r < 8; ++r) tmax[r] = fmaxf(s0[r], s1[r]);
    #pragma unroll
    for (int off = 1; off < 16; off <<= 1) {
      #pragma unroll
      for (int r = 0; r < 8; ++r)
        tmax[r] = fmaxf(tmax[r], __shfl_xor(tmax[r], off, 32));
    }
    float scl[8];
    #pragma unroll
    for (int r = 0; r < 8; ++r) {
      float nm = fmaxf(mst[r], tmax[r]);
      scl[r] = __expf(mst[r] - nm);
      mst[r] = nm;
    }
    float tsum[8];
    #pragma unroll
    for (int r = 0; r < 8; ++r) {
      s0[r] = __expf(s0[r] - mst[r]);
      s1[r] = __expf(s1[r] - mst[r]);
      tsum[r] = s0[r] + s1[r];
    }
    #pragma unroll
    for (int off = 1; off < 16; off <<= 1) {
      #pragma unroll
      for (int r = 0; r < 8; ++r)
        tsum[r] += __shfl_xor(tsum[r], off, 32);
    }
    #pragma unroll
    for (int r = 0; r < 8; ++r) {
      lst[r] = lst[r] * scl[r] + tsum[r];
      o0[r] *= scl[r]; o1[r] *= scl[r]; o2[r] *= scl[r];
    }
    // C-layout (col per lane) -> A-layout (row per lane) via 16x32 bf16 LDS tile
    #pragma unroll
    for (int r = 0; r < 8; ++r) {
      pls[(r + 8 * hh) * 32 + n16]      = (bf16)s0[r];
      pls[(r + 8 * hh) * 32 + 16 + n16] = (bf16)s1[r];
    }
    __syncthreads();
    bf16x16 ap;
    {
      const bf16* p = pls + (lane & 15) * 32 + 8 * hh;
      FragCast u;
      u.q[0] = *(const i32x4*)(p);
      u.q[1] = *(const i32x4*)(p + 16);
      ap = u.v;
    }
    __syncthreads();
    // O += P(16x32) @ Vhat(32x48)
    bf16x16 bv0 = load_frag_b(V, NSEQ, 0,  j0);
    bf16x16 bv1 = load_frag_b(V, NSEQ, 16, j0);
    bf16x16 bv2 = load_frag_b(V, NSEQ, 32, j0);
    o0 = WMMA_BF16(ap, bv0, o0);
    o1 = WMMA_BF16(ap, bv1, o1);
    o2 = WMMA_BF16(ap, bv2, o2);
  }
  // normalize + scatter into concat layout [b*n][352]: [h*32+d | 256 + h*12 + pd]
  #pragma unroll
  for (int r = 0; r < 8; ++r) {
    float inv = 1.0f / lst[r];
    int i = i0 + r + 8 * hh;
    size_t rowb = ((size_t)bidx * NSEQ + i) * CAT_LD;
    attcat[rowb + hd * 32 + n16]      = (bf16)(o0[r] * inv);
    attcat[rowb + hd * 32 + 16 + n16] = (bf16)(o1[r] * inv);
    if (n16 < 12) attcat[rowb + 256 + hd * 12 + n16] = (bf16)(o2[r] * inv);
  }
}

/* ---------------- output GEMM: concat(4096x352) @ Wo(352x256) + bo ---------------- */

__global__ __launch_bounds__(32) void k_gemm_out(
    const bf16* __restrict__ attcat, const bf16* __restrict__ WoT,
    const float* bo, float* out) {
  const int row0 = blockIdx.x * 16;
  const int col0 = blockIdx.y * 32;
  const int lane = threadIdx.x & 31;
  const int hh = lane >> 4, n16 = lane & 15;
  f32x8 c0 = {0,0,0,0,0,0,0,0}, c1 = {0,0,0,0,0,0,0,0};
  for (int k0 = 0; k0 < CAT_LD; k0 += 32) {
    bf16x16 a  = load_frag_a(attcat, CAT_LD, row0, k0);
    bf16x16 b0 = load_frag_b(WoT, CAT_LD, col0, k0);
    bf16x16 b1 = load_frag_b(WoT, CAT_LD, col0 + 16, k0);
    c0 = WMMA_BF16(a, b0, c0);
    c1 = WMMA_BF16(a, b1, c1);
  }
  #pragma unroll
  for (int t = 0; t < 2; ++t) {
    f32x8 c = t ? c1 : c0;
    int col = col0 + 16 * t + n16;
    float bb = bo[col];
    #pragma unroll
    for (int r = 0; r < 8; ++r) {
      int g = row0 + r + 8 * hh;
      out[(size_t)g * 256 + col] = c[r] + bb;
    }
  }
}

/* ---------------- launch ---------------- */

extern "C" void kernel_launch(void* const* d_in, const int* in_sizes, int n_in,
                              void* d_out, int out_size, void* d_ws, size_t ws_size,
                              hipStream_t stream) {
  (void)in_sizes; (void)n_in; (void)out_size; (void)ws_size;
  const float* features = (const float*)d_in[0];
  const float* coords   = (const float*)d_in[1];
  const unsigned char* mask = (const unsigned char*)d_in[2];
  const float* Wq  = (const float*)d_in[3];
  const float* bq  = (const float*)d_in[4];
  const float* Wk  = (const float*)d_in[5];
  const float* bk  = (const float*)d_in[6];
  const float* Wv  = (const float*)d_in[7];
  const float* bv  = (const float*)d_in[8];
  const float* Wqp = (const float*)d_in[9];
  const float* bqp = (const float*)d_in[10];
  const float* Wkp = (const float*)d_in[11];
  const float* bkp = (const float*)d_in[12];
  const float* Wvp = (const float*)d_in[13];
  const float* bvp = (const float*)d_in[14];
  const float* Wo  = (const float*)d_in[15];
  const float* bo  = (const float*)d_in[16];
  const float* w_c = (const float*)d_in[17];
  const float* w_l = (const float*)d_in[18];
  float* out = (float*)d_out;

  char* ws = (char*)d_ws;
  size_t off = 0;
  auto alloc = [&](size_t bytes) -> char* {
    char* p = ws + off;
    off += (bytes + 255) & ~(size_t)255;
    return p;
  };
  bf16* Fbf    = (bf16*)alloc((size_t)ROWS * 256 * 2);
  bf16* WT     = (bf16*)alloc((size_t)NCOLS_PROJ * 256 * 2);
  bf16* WoT    = (bf16*)alloc((size_t)256 * CAT_LD * 2);
  bf16* Qhat   = (bf16*)alloc((size_t)BATCH * NHEADS * NSEQ * QK_LD * 2);
  bf16* Khat   = (bf16*)alloc((size_t)BATCH * NHEADS * NSEQ * QK_LD * 2);
  bf16* VhatT  = (bf16*)alloc((size_t)BATCH * NHEADS * V_CD * NSEQ * 2);
  float* biasK = (float*)alloc((size_t)BATCH * NHEADS * NSEQ * 4);
  bf16* attcat = (bf16*)alloc((size_t)ROWS * CAT_LD * 2);

  // zero pad columns (K padding 44->64 must be exact zero)
  hipMemsetAsync(Qhat,  0, (size_t)BATCH * NHEADS * NSEQ * QK_LD * 2, stream);
  hipMemsetAsync(Khat,  0, (size_t)BATCH * NHEADS * NSEQ * QK_LD * 2, stream);
  hipMemsetAsync(VhatT, 0, (size_t)BATCH * NHEADS * V_CD * NSEQ * 2, stream);

  k_prep_feat<<<(ROWS * 256 + 255) / 256, 256, 0, stream>>>(features, Fbf, ROWS * 256);
  k_prep_wcat<<<(NCOLS_PROJ * 256 + 255) / 256, 256, 0, stream>>>(Wq, Wk, Wv, Wqp, Wkp, Wvp, WT);
  k_prep_woT<<<(256 * CAT_LD + 255) / 256, 256, 0, stream>>>(Wo, WoT);

  k_gemm_proj<<<dim3(ROWS / 16, NCOLS_PROJ / 32), 32, 0, stream>>>(
      Fbf, WT, bq, bk, bv, bqp, bkp, bvp, coords, w_l, w_c, Qhat, Khat, VhatT);

  k_kk_bias<<<(BATCH * NHEADS * NSEQ + 255) / 256, 256, 0, stream>>>(Khat, w_c, biasK);

  k_attn<<<dim3(BATCH * NHEADS, NSEQ / 16), 32, 0, stream>>>(
      Qhat, Khat, VhatT, biasK, mask, attcat);

  k_gemm_out<<<dim3(ROWS / 16, 256 / 32), 32, 0, stream>>>(attcat, WoT, bo, out);
}
